// BaselineRNN_78374563217746
// MI455X (gfx1250) — matrix operand order repaired and checked by reading, not compiled, not run on hardware
//
#include <hip/hip_runtime.h>
#include <hip/hip_bf16.h>
#include <stdint.h>

// ---------------------------------------------------------------------------
// Elman RNN scan: h_t = tanh([x_t; h_{t-1}] @ W_i2h^T + b), out = h_S @ W_h2o^T + b.
//
// - Recurrence is independent per batch row -> 8 persistent workgroups, each
//   owns 16 batch rows; hidden state lives in LDS (bf16 ping-pong), no grid
//   sync ever.
// - 512 threads = 16 waves per WG; each wave owns two 16-wide H tiles, so
//   every A-operand LDS load feeds two v_wmma_f32_16x16x32_bf16.
// - LDS rows padded by 8 bf16 so the 16 lanes of a half-wave spread across
//   all 64 banks (unpadded stride was = 0 mod 64 -> 16-way conflicts).
// - Weights converted f32->bf16 once into d_ws (~1 MB) and re-streamed from
//   the 192MB L2 every step. An opaque zero *offset* (asm "+s") blocks LLVM
//   from hoisting the invariant weight loads (which spilled to scratch),
//   while keeping addrspace(1) inference so they lower as global_load_b128,
//   not flat_load (flat would also burn DScnt and contend with LDS).
// ---------------------------------------------------------------------------

typedef __attribute__((ext_vector_type(16))) __bf16 v16bf;
typedef __attribute__((ext_vector_type(8)))  float  v8f;

#define RNN_B 128
#define RNN_S 1024
#define RNN_I 256
#define RNN_H 512
#define RNN_O 256
#define RNN_K (RNN_I + RNN_H)   // 768 fused contraction dim
#define XPAD  (RNN_I + 8)       // LDS row strides padded: stride % 64 dwords == 4
#define HPAD  (RNN_H + 8)

union BfVec { v16bf v; float4 f[2]; };

__device__ __forceinline__ v16bf load2x16B(const __hip_bfloat16* p0,
                                           const __hip_bfloat16* p1) {
  BfVec u;
  u.f[0] = *reinterpret_cast<const float4*>(p0);
  u.f[1] = *reinterpret_cast<const float4*>(p1);
  return u.v;
}

__global__ void convert_weights_kernel(const float* __restrict__ Wi2h,
                                       const float* __restrict__ Wh2o,
                                       __hip_bfloat16* __restrict__ dst) {
  const int n1  = RNN_H * RNN_K;                       // 393216
  const int idx = blockIdx.x * blockDim.x + threadIdx.x;
  if (idx < n1) dst[idx] = __float2bfloat16(Wi2h[idx]);
  else          dst[idx] = __float2bfloat16(Wh2o[idx - n1]);
}

__global__ __launch_bounds__(512, 1)
void rnn_scan_kernel(const float* __restrict__ seq,     // (B,S,I) f32
                     const float* __restrict__ b_i2h,   // (H)
                     const float* __restrict__ b_h2o,   // (O)
                     const __hip_bfloat16* __restrict__ w_in,   // (H, K) bf16
                     const __hip_bfloat16* __restrict__ w_out,  // (O, H) bf16
                     float* __restrict__ out)           // [B*O output | B*H hidden]
{
  __shared__ __align__(16) __hip_bfloat16 xbuf[16][XPAD];       // ~8.3 KB
  __shared__ __align__(16) __hip_bfloat16 hbuf[2][16][HPAD];    // ~32.5 KB

  const int wg   = blockIdx.x;        // batch-row block: rows [wg*16, wg*16+16)
  const int tid  = threadIdx.x;
  const int wave = tid >> 5;          // 0..15 (wave32)
  const int lane = tid & 31;
  const int g    = lane >> 4;         // wave half -> K split of WMMA operands
  const int lm   = lane & 15;         // A row / B,C column within 16x16 tile

  // h0 = 0 (zero the padding too; harmless)
  for (int i = tid; i < 16 * HPAD; i += 512)
    (&hbuf[0][0][0])[i] = __float2bfloat16(0.0f);

  const int n0 = wave * 32;           // each wave owns two adjacent 16-wide H tiles
  const size_t woff0 = (size_t)(n0 + lm) * RNN_K;       // per-lane W rows (elements)
  const size_t woff1 = (size_t)(n0 + 16 + lm) * RNN_K;
  const float bias0 = b_i2h[n0 + lm];
  const float bias1 = b_i2h[n0 + 16 + lm];

  const int xr  = tid >> 5;           // x staging: one row per 32-thread group
  const int xi0 = (tid & 31) * 8;     // 8 f32 elements per thread

  int cur = 0;
  __syncthreads();

  #pragma unroll 1
  for (int t = 0; t < RNN_S; ++t) {
    // Opaque zero offset: blocks hoisting of the loop-invariant weight loads
    // (register spill) while keeping global address-space inference intact.
    int zoff = 0;
    asm volatile("" : "+s"(zoff));
    const __hip_bfloat16* wrow0 = w_in + woff0 + zoff;
    const __hip_bfloat16* wrow1 = w_in + woff1 + zoff;

    // ---- stage x_t block (16 x 256) f32 -> bf16 into LDS ----
    {
      const float4* src = reinterpret_cast<const float4*>(
          seq + ((size_t)(wg * 16 + xr) * RNN_S + t) * RNN_I + xi0);
      float4 a = src[0], b = src[1];
      __hip_bfloat16* dst = &xbuf[xr][xi0];
      dst[0] = __float2bfloat16(a.x); dst[1] = __float2bfloat16(a.y);
      dst[2] = __float2bfloat16(a.z); dst[3] = __float2bfloat16(a.w);
      dst[4] = __float2bfloat16(b.x); dst[5] = __float2bfloat16(b.y);
      dst[6] = __float2bfloat16(b.z); dst[7] = __float2bfloat16(b.w);
    }
    __syncthreads();

    v8f c0, c1;
    #pragma unroll
    for (int i = 0; i < 8; ++i) { c0[i] = bias0; c1[i] = bias1; }

    // ---- x @ Wx : K = 0..255 (A from LDS, B rows streamed from L2) ----
    const __hip_bfloat16* arow = &xbuf[lm][0];
    #pragma unroll
    for (int kb = 0; kb < RNN_I / 32; ++kb) {
      const __hip_bfloat16* ab = arow + kb * 32;
      const __hip_bfloat16* w0 = wrow0 + kb * 32 + g * 16;
      const __hip_bfloat16* w1 = wrow1 + kb * 32 + g * 16;
      v16bf a  = load2x16B(ab + g * 16, ab + g * 16 + 8);
      v16bf b0 = load2x16B(w0, w0 + 8);
      v16bf b1 = load2x16B(w1, w1 + 8);
      c0 = __builtin_amdgcn_wmma_f32_16x16x32_bf16(false, a, false, b0, (short)0, c0, false, false);
      c1 = __builtin_amdgcn_wmma_f32_16x16x32_bf16(false, a, false, b1, (short)0, c1, false, false);
    }
    // ---- h @ Wh : K = 256..767 ----
    const __hip_bfloat16* hrow = &hbuf[cur][lm][0];
    #pragma unroll
    for (int kb = 0; kb < RNN_H / 32; ++kb) {
      const __hip_bfloat16* ab = hrow + kb * 32;
      const __hip_bfloat16* w0 = wrow0 + RNN_I + kb * 32 + g * 16;
      const __hip_bfloat16* w1 = wrow1 + RNN_I + kb * 32 + g * 16;
      v16bf a  = load2x16B(ab + g * 16, ab + g * 16 + 8);
      v16bf b0 = load2x16B(w0, w0 + 8);
      v16bf b1 = load2x16B(w1, w1 + 8);
      c0 = __builtin_amdgcn_wmma_f32_16x16x32_bf16(false, a, false, b0, (short)0, c0, false, false);
      c1 = __builtin_amdgcn_wmma_f32_16x16x32_bf16(false, a, false, b1, (short)0, c1, false, false);
    }

    // ---- h_new = tanh(acc); C layout: elem i -> row (i + 8*g), col lm ----
    const int nxt = cur ^ 1;
    #pragma unroll
    for (int i = 0; i < 8; ++i) {
      float v0 = tanhf(c0[i]);
      float v1 = tanhf(c1[i]);
      const int m = i + 8 * g;
      hbuf[nxt][m][n0 + lm]      = __float2bfloat16(v0);
      hbuf[nxt][m][n0 + 16 + lm] = __float2bfloat16(v1);
      if (t == RNN_S - 1) {                      // final hidden state, f32
        float* hid = out + RNN_B * RNN_O;
        hid[(size_t)(wg * 16 + m) * RNN_H + n0 + lm]      = v0;
        hid[(size_t)(wg * 16 + m) * RNN_H + n0 + 16 + lm] = v1;
      }
    }
    __syncthreads();
    cur = nxt;
  }

  // ---- output projection: wave == O-tile (16 tiles of 16 cols), K = 512 ----
  {
    v8f co;
    const float biaso = b_h2o[wave * 16 + lm];
    #pragma unroll
    for (int i = 0; i < 8; ++i) co[i] = biaso;
    const __hip_bfloat16* orow = w_out + (size_t)(wave * 16 + lm) * RNN_H;
    const __hip_bfloat16* hrow = &hbuf[cur][lm][0];
    #pragma unroll
    for (int kb = 0; kb < RNN_H / 32; ++kb) {
      const __hip_bfloat16* ab = hrow + kb * 32;
      const __hip_bfloat16* wb = orow + kb * 32 + g * 16;
      v16bf a = load2x16B(ab + g * 16, ab + g * 16 + 8);
      v16bf b = load2x16B(wb, wb + 8);
      co = __builtin_amdgcn_wmma_f32_16x16x32_bf16(false, a, false, b, (short)0, co, false, false);
    }
    #pragma unroll
    for (int i = 0; i < 8; ++i)
      out[(size_t)(wg * 16 + i + 8 * g) * RNN_O + wave * 16 + lm] = co[i];
  }
}

extern "C" void kernel_launch(void* const* d_in, const int* in_sizes, int n_in,
                              void* d_out, int out_size, void* d_ws, size_t ws_size,
                              hipStream_t stream) {
  const float* seq  = (const float*)d_in[0];   // (B,S,I)
  const float* Wi2h = (const float*)d_in[1];   // (H, I+H)
  const float* bi2h = (const float*)d_in[2];   // (H)
  const float* Wh2o = (const float*)d_in[3];   // (O, H)
  const float* bh2o = (const float*)d_in[4];   // (O)
  float* out = (float*)d_out;                  // [B*O output | B*H hidden]

  __hip_bfloat16* w_in  = (__hip_bfloat16*)d_ws;           // H*K bf16
  __hip_bfloat16* w_out = w_in + (size_t)RNN_H * RNN_K;    // O*H bf16

  const int nconv = RNN_H * RNN_K + RNN_O * RNN_H;         // 524288 = 2048*256
  convert_weights_kernel<<<dim3(nconv / 256), dim3(256), 0, stream>>>(Wi2h, Wh2o, w_in);
  rnn_scan_kernel<<<dim3(RNN_B / 16), dim3(512), 0, stream>>>(seq, bi2h, bh2o,
                                                              w_in, w_out, out);
}
